// MultiHeadLocalAttention_3728031613534
// MI455X (gfx1250) — matrix operand order repaired and checked by reading, not compiled, not run on hardware
//
#include <hip/hip_runtime.h>

// ---------------------------------------------------------------------------
// MI455X (gfx1250) multi-head block-local attention, bf16 WMMA path.
//   y = softmax((xWq^T+bq)(xWk^T+bk)^T / sqrt(dk)) (xWv^T+bv)  -> @Wo^T + bo
// Strategy: stage x and weights as bf16 (activation tensor = 64MB, fits the
// 192MB L2), run all GEMMs + attention through v_wmma_f32_16x16x32_bf16
// (fp32 accumulate). GEMM LDS tiles are staged with CDNA5 async-to-LDS copies
// (ASYNCcnt) and double-buffered so staging overlaps the WMMA pipe.
// ---------------------------------------------------------------------------

typedef __attribute__((ext_vector_type(16))) __bf16 v16bf;
typedef __attribute__((ext_vector_type(8)))  float  v8f;

struct alignas(16) U128 { unsigned int x, y, z, w; };

union FragBF {
    U128  q[2];   // two 16B chunks, element order matches v16bf lanes 0..7 / 8..15
    v16bf v;
};

__device__ __forceinline__ unsigned short f32_to_bf16(float f) {
    unsigned int u = __float_as_uint(f);
    unsigned int r = u + 0x7FFFu + ((u >> 16) & 1u);   // round-to-nearest-even
    return (unsigned short)(r >> 16);
}

__device__ __forceinline__ void store_val(float* C, size_t i, float v) { C[i] = v; }
__device__ __forceinline__ void store_val(unsigned short* C, size_t i, float v) { C[i] = f32_to_bf16(v); }

// CDNA5 async global->LDS copy (16B per lane), tracked by ASYNCcnt.
// GV mode: %0 = per-lane LDS byte address (VGPR), %1 = per-lane 64-bit global
// address (VGPR pair), no SADDR.
__device__ __forceinline__ void async_load_b128(unsigned lds_off, const void* gptr) {
    asm volatile("global_load_async_to_lds_b128 %0, %1, off"
                 :: "v"(lds_off), "v"(gptr) : "memory");
}
__device__ __forceinline__ void wait_async0() {
    asm volatile("s_wait_asynccnt 0" ::: "memory");
}
// Generic LDS pointer -> 32-bit LDS address (ISA: LDS aperture low 32 bits).
__device__ __forceinline__ unsigned lds_addr32(const void* p) {
    return (unsigned)(uintptr_t)p;
}

// ---------------------------------------------------------------------------
// fp32 -> bf16 bits, grid-stride
// ---------------------------------------------------------------------------
__global__ void f32_to_bf16_kernel(const float* __restrict__ in,
                                   unsigned short* __restrict__ out, int n) {
    int i = blockIdx.x * blockDim.x + threadIdx.x;
    int stride = gridDim.x * blockDim.x;
    for (; i < n; i += stride) out[i] = f32_to_bf16(in[i]);
}

// ---------------------------------------------------------------------------
// C[M,N] = A[M,K] * Bw[N,K]^T + bias[N]
// A, Bw are bf16 bits; C is fp32 or bf16 bits.
// Block tile 128x128, K-step 32 (one WMMA depth). 8 waves: 4 along M x 2
// along N, each wave owns a 32x64 sub-tile = 2x4 WMMA accumulators.
// LDS tiles double-buffered, filled with global_load_async_to_lds_b128.
// ---------------------------------------------------------------------------
template <typename OutT>
__global__ __launch_bounds__(256)
void gemm_bf16_wmma(const unsigned short* __restrict__ A,
                    const unsigned short* __restrict__ Bw,
                    const float* __restrict__ bias,
                    OutT* __restrict__ C,
                    int M, int N, int K)
{
    constexpr int BM = 128, BN = 128, BK = 32;
    constexpr int LDA = BK + 16;                     // 48 ushorts -> 96B rows (16B aligned)
    __shared__ unsigned short As[2][BM * LDA];       // 2 x 12 KB
    __shared__ unsigned short Bs[2][BN * LDA];       // 2 x 12 KB

    const int tid  = threadIdx.x;
    const int lane = tid & 31;
    const int wave = tid >> 5;
    const int lm   = lane & 15;
    const int hi   = lane >> 4;

    const int row0 = blockIdx.y * BM;
    const int col0 = blockIdx.x * BN;
    const int wm   = (wave >> 1) * 32;   // wave's M offset in tile
    const int wn   = (wave & 1) * 64;    // wave's N offset in tile

    v8f acc[2][4];
    for (int i = 0; i < 2; ++i)
        for (int j = 0; j < 4; ++j)
            for (int e = 0; e < 8; ++e) acc[i][j][e] = 0.0f;

    // staging: thread t copies 16 ushorts (32B = 2 async b128) of row t>>1
    // at col (t&1)*16, for both A and B tiles.
    const int srow = tid >> 1;
    const int scol = (tid & 1) * 16;
    const unsigned short* gA0 = A  + (size_t)(row0 + srow) * K + scol;
    const unsigned short* gB0 = Bw + (size_t)(col0 + srow) * K + scol;
    const int soff = srow * LDA + scol;

    // prologue: stage first tile
    {
        async_load_b128(lds_addr32(&As[0][soff]),     gA0);
        async_load_b128(lds_addr32(&As[0][soff + 8]), gA0 + 8);
        async_load_b128(lds_addr32(&Bs[0][soff]),     gB0);
        async_load_b128(lds_addr32(&Bs[0][soff + 8]), gB0 + 8);
    }
    wait_async0();
    __syncthreads();

    for (int k0 = 0; k0 < K; k0 += BK) {
        const int buf = (k0 / BK) & 1;
        // issue async copies for the next tile into the other buffer
        if (k0 + BK < K) {
            const int nb = buf ^ 1;
            async_load_b128(lds_addr32(&As[nb][soff]),     gA0 + k0 + BK);
            async_load_b128(lds_addr32(&As[nb][soff + 8]), gA0 + k0 + BK + 8);
            async_load_b128(lds_addr32(&Bs[nb][soff]),     gB0 + k0 + BK);
            async_load_b128(lds_addr32(&Bs[nb][soff + 8]), gB0 + k0 + BK + 8);
        }

        // A fragments (lanes 0-15 hold K 0..15, lanes 16-31 hold K 16..31)
        FragBF af[2];
        #pragma unroll
        for (int i = 0; i < 2; ++i) {
            const U128* pv = reinterpret_cast<const U128*>(&As[buf][(wm + i * 16 + lm) * LDA + hi * 16]);
            af[i].q[0] = pv[0];   // K = hi*16 + 0..7
            af[i].q[1] = pv[1];   // K = hi*16 + 8..15
        }
        // B fragments (lanes 0-15: K 0..15, lanes 16-31: K 16..31, N = lane&15)
        FragBF bf_[4];
        #pragma unroll
        for (int j = 0; j < 4; ++j) {
            const U128* pv = reinterpret_cast<const U128*>(&Bs[buf][(wn + j * 16 + lm) * LDA + hi * 16]);
            bf_[j].q[0] = pv[0];
            bf_[j].q[1] = pv[1];
        }
        #pragma unroll
        for (int i = 0; i < 2; ++i)
            #pragma unroll
            for (int j = 0; j < 4; ++j)
                acc[i][j] = __builtin_amdgcn_wmma_f32_16x16x32_bf16(
                    false, af[i].v, false, bf_[j].v, (short)0, acc[i][j], false, false);

        // own async copies landed in LDS, then block-wide barrier before swap
        wait_async0();
        __syncthreads();
    }

    // Epilogue: C layout M = r + 8*hi, N = lane&15
    for (int i = 0; i < 2; ++i) {
        const int rbase = row0 + wm + i * 16 + hi * 8;
        for (int j = 0; j < 4; ++j) {
            const int col = col0 + wn + j * 16 + lm;
            const float bv = bias[col];
            #pragma unroll
            for (int r = 0; r < 8; ++r)
                store_val(C, (size_t)(rbase + r) * N + col, acc[i][j][r] + bv);
        }
    }
}

// ---------------------------------------------------------------------------
// Block-local attention. One block per (window, head): Q,K,V tiles 128x64 bf16.
// scores = QK^T/8 -> softmax (in-register, shfl_xor over 16-lane halves)
// P bf16 through LDS (C-layout -> A-layout reshape), V transposed in LDS,
// O = P V, scaled by 1/rowsum, written bf16.
// ---------------------------------------------------------------------------
__global__ __launch_bounds__(256)
void local_attn_wmma(const unsigned short* __restrict__ Qb,
                     const unsigned short* __restrict__ Kb,
                     const unsigned short* __restrict__ Vb,
                     unsigned short* __restrict__ Ob)
{
    constexpr int D = 1024, W = 128, DK = 64;
    constexpr int LV = 136;   // Vt row stride (272B, 16B aligned)
    constexpr int LP = 136;   // Ps row stride
    __shared__ unsigned short Vt[DK * LV];        // ~17 KB  V transposed [dchan][key]
    __shared__ unsigned short Ps[8 * 16 * LP];    // ~34 KB  per-wave P rows

    const int win = blockIdx.x;        // 0..255 (windows never cross batch: 8192%128==0)
    const int h   = blockIdx.y;        // 0..15
    const int t0  = win * W;
    const int hc  = h * DK;

    const int tid  = threadIdx.x;
    const int lane = tid & 31;
    const int wave = tid >> 5;
    const int lm   = lane & 15;
    const int hi   = lane >> 4;

    // stage V transposed: thread t handles key=t>>1, 32 channels at (t&1)*32
    {
        const int key = tid >> 1;
        const int cb  = (tid & 1) * 32;
        const unsigned short* src = Vb + (size_t)(t0 + key) * D + hc + cb;
        #pragma unroll
        for (int c = 0; c < 32; ++c)
            Vt[(cb + c) * LV + key] = src[c];
    }
    __syncthreads();

    // Q fragments straight from global (wave owns query rows wave*16..+15)
    const int qrow = t0 + wave * 16 + lm;
    FragBF aq[2];
    #pragma unroll
    for (int ks = 0; ks < 2; ++ks) {
        const U128* pv = reinterpret_cast<const U128*>(Qb + (size_t)qrow * D + hc + ks * 32 + hi * 16);
        aq[ks].q[0] = pv[0];
        aq[ks].q[1] = pv[1];
    }

    // scores: 16x128 stripe per wave = 8 N-tiles x 2 K-steps
    v8f sc[8];
    for (int j = 0; j < 8; ++j)
        for (int e = 0; e < 8; ++e) sc[j][e] = 0.0f;
    for (int j = 0; j < 8; ++j) {
        #pragma unroll
        for (int ks = 0; ks < 2; ++ks) {
            FragBF bk;   // B fragment from K (rows of K are contiguous along dk)
            const U128* pv = reinterpret_cast<const U128*>(
                Kb + (size_t)(t0 + j * 16 + lm) * D + hc + ks * 32 + hi * 16);
            bk.q[0] = pv[0]; bk.q[1] = pv[1];
            sc[j] = __builtin_amdgcn_wmma_f32_16x16x32_bf16(
                false, aq[ks].v, false, bk.v, (short)0, sc[j], false, false);
        }
    }

    // softmax over rows M = r + 8*hi (values spread across 16-lane half)
    const float scale = 0.125f;   // 1/sqrt(64)
    float rsum[8];
    #pragma unroll
    for (int r = 0; r < 8; ++r) {
        float m = -3.4e38f;
        for (int j = 0; j < 8; ++j) m = fmaxf(m, sc[j][r]);
        for (int s = 1; s < 16; s <<= 1) m = fmaxf(m, __shfl_xor(m, s, 32));
        float sum = 0.f;
        for (int j = 0; j < 8; ++j) {
            float e = __expf((sc[j][r] - m) * scale);
            sc[j][r] = e;
            sum += e;
        }
        for (int s = 1; s < 16; s <<= 1) sum += __shfl_xor(sum, s, 32);
        rsum[r] = sum;
    }

    // reshape P: C-layout (M=r+8*hi, col=lane&15) -> LDS row-major for A-frags
    unsigned short* pw = Ps + wave * 16 * LP;
    for (int j = 0; j < 8; ++j)
        #pragma unroll
        for (int r = 0; r < 8; ++r)
            pw[(r + 8 * hi) * LP + j * 16 + lm] = f32_to_bf16(sc[j][r]);
    __syncthreads();

    // O = P @ V : 4 N-tiles (dk=64), 4 K-steps (keys=128)
    v8f oacc[4];
    for (int j = 0; j < 4; ++j)
        for (int e = 0; e < 8; ++e) oacc[j][e] = 0.0f;
    for (int ks = 0; ks < 4; ++ks) {
        FragBF ap;
        const U128* pv = reinterpret_cast<const U128*>(pw + lm * LP + ks * 32 + hi * 16);
        ap.q[0] = pv[0];
        ap.q[1] = pv[1];
        #pragma unroll
        for (int j = 0; j < 4; ++j) {
            FragBF bv;
            const U128* qv = reinterpret_cast<const U128*>(Vt + (j * 16 + lm) * LV + ks * 32 + hi * 16);
            bv.q[0] = qv[0]; bv.q[1] = qv[1];
            oacc[j] = __builtin_amdgcn_wmma_f32_16x16x32_bf16(
                false, ap.v, false, bv.v, (short)0, oacc[j], false, false);
        }
    }

    // normalize + store bf16 (row M = r + 8*hi matches rsum layout)
    const int orow = t0 + wave * 16;
    for (int j = 0; j < 4; ++j) {
        const int col = hc + j * 16 + lm;
        #pragma unroll
        for (int r = 0; r < 8; ++r) {
            float v = oacc[j][r] / rsum[r];
            Ob[(size_t)(orow + r + 8 * hi) * D + col] = f32_to_bf16(v);
        }
    }
}

// ---------------------------------------------------------------------------
// Launch
// ---------------------------------------------------------------------------
extern "C" void kernel_launch(void* const* d_in, const int* in_sizes, int n_in,
                              void* d_out, int out_size, void* d_ws, size_t ws_size,
                              hipStream_t stream) {
    const float* x  = (const float*)d_in[0];
    const float* Wq = (const float*)d_in[1];
    const float* bq = (const float*)d_in[2];
    const float* Wk = (const float*)d_in[3];
    const float* bk = (const float*)d_in[4];
    const float* Wv = (const float*)d_in[5];
    const float* bv = (const float*)d_in[6];
    const float* Wo = (const float*)d_in[7];
    const float* bo = (const float*)d_in[8];
    float* out = (float*)d_out;

    constexpr size_t NTOK = 4 * 8192;   // 32768 tokens
    constexpr size_t Dm   = 1024;

    // Workspace layout (bf16 bits), ~344 MB total
    unsigned short* base = (unsigned short*)d_ws;
    unsigned short* xb  = base;
    unsigned short* Wqb = xb  + NTOK * Dm;
    unsigned short* Wkb = Wqb + Dm * Dm;
    unsigned short* Wvb = Wkb + Dm * Dm;
    unsigned short* Wob = Wvb + Dm * Dm;
    unsigned short* Qb  = Wob + Dm * Dm;
    unsigned short* Kb  = Qb  + NTOK * Dm;
    unsigned short* Vb  = Kb  + NTOK * Dm;
    unsigned short* Ob  = Vb  + NTOK * Dm;

    // 1) fp32 -> bf16 staging
    f32_to_bf16_kernel<<<4096, 256, 0, stream>>>(x,  xb,  (int)(NTOK * Dm));
    f32_to_bf16_kernel<<<1024, 256, 0, stream>>>(Wq, Wqb, (int)(Dm * Dm));
    f32_to_bf16_kernel<<<1024, 256, 0, stream>>>(Wk, Wkb, (int)(Dm * Dm));
    f32_to_bf16_kernel<<<1024, 256, 0, stream>>>(Wv, Wvb, (int)(Dm * Dm));
    f32_to_bf16_kernel<<<1024, 256, 0, stream>>>(Wo, Wob, (int)(Dm * Dm));

    // 2) Q/K/V projections (bf16 out)
    dim3 gg(Dm / 128, NTOK / 128);   // (8, 256)
    gemm_bf16_wmma<unsigned short><<<gg, 256, 0, stream>>>(xb, Wqb, bq, Qb, (int)NTOK, (int)Dm, (int)Dm);
    gemm_bf16_wmma<unsigned short><<<gg, 256, 0, stream>>>(xb, Wkb, bk, Kb, (int)NTOK, (int)Dm, (int)Dm);
    gemm_bf16_wmma<unsigned short><<<gg, 256, 0, stream>>>(xb, Wvb, bv, Vb, (int)NTOK, (int)Dm, (int)Dm);

    // 3) windowed attention: 256 windows x 16 heads
    local_attn_wmma<<<dim3(256, 16), 256, 0, stream>>>(Qb, Kb, Vb, Ob);

    // 4) output projection (fp32 out + bias)
    gemm_bf16_wmma<float><<<gg, 256, 0, stream>>>(Ob, Wob, bo, out, (int)NTOK, (int)Dm, (int)Dm);
}
